// FAM_model_5634997092903
// MI455X (gfx1250) — matrix operand order repaired and checked by reading, not compiled, not run on hardware
//
#include <hip/hip_runtime.h>
#include <hip/hip_bf16.h>

typedef __attribute__((ext_vector_type(16))) _Float16 v16h;
typedef __attribute__((ext_vector_type(8)))  float    v8f;
typedef __attribute__((ext_vector_type(4)))  float    f4;

#define Bq   2
#define C    64
#define S    128
#define HW   (S*S)            // 16384
#define NPIX (Bq*HW)          // 32768
#define NTOT (Bq*C*HW)        // 2097152

// ---------------------------------------------------------------------------
// 1) wsum[c,hw] = sum_i weights1[c,i,hw]  (real & imag).  512 MB streamed —
//    the roofline-dominant pass.  float4 non-temporal loads + prefetch.
// ---------------------------------------------------------------------------
__global__ void wsum_kernel(const f4* __restrict__ wr, const f4* __restrict__ wi,
                            f4* __restrict__ sr, f4* __restrict__ si) {
    int idx = blockIdx.x * blockDim.x + threadIdx.x;   // 0 .. 64*4096-1
    if (idx >= 64 * 4096) return;
    int c = idx >> 12;            // CIN channel
    int q = idx & 4095;           // float4 group within hw
    long base = ((long)c * 64) * 4096 + q;             // [c][i][hw/4]
    f4 ar = {0.f, 0.f, 0.f, 0.f}, ai = {0.f, 0.f, 0.f, 0.f};
    for (int i = 0; i < 64; ++i) {
        if (i + 8 < 64) {
            __builtin_prefetch(&wr[base + (long)(i + 8) * 4096], 0, 0);
            __builtin_prefetch(&wi[base + (long)(i + 8) * 4096], 0, 0);
        }
        f4 r = __builtin_nontemporal_load(&wr[base + (long)i * 4096]);
        f4 m = __builtin_nontemporal_load(&wi[base + (long)i * 4096]);
        ar += r; ai += m;
    }
    sr[idx] = ar; si[idx] = ai;
}

// ---------------------------------------------------------------------------
// 2) 64x64 GEMM over pixels via V_WMMA_F32_16X16X32_F16.
//    Y[b,m,hw] = sum_c W[m,c] * X[b,c,hw] (+ bias[m]).
//    One wave per 16x16 output tile; K=64 -> two WMMA ops.
// ---------------------------------------------------------------------------
__global__ __launch_bounds__(32)
void gemm64_wmma(const float* __restrict__ W, const float* __restrict__ bias,
                 const float* __restrict__ X, float* __restrict__ Y) {
    int lane = threadIdx.x;
    int nl = lane & 15, hi = lane >> 4;
    int mt = blockIdx.y;                 // 0..3   (64 rows / 16)
    int nt = blockIdx.x;                 // 0..2047 (32768 pixels / 16)
    int m  = (mt << 4) + nl;             // A matrix row (lanes 0-15 & 16-31 both M=0..15)
    int p  = (nt << 4) + nl;             // output pixel column
    int bimg = p >> 14;                  // batch
    int hw   = p & 16383;
    long xbase = (long)bimg * (64L * HW) + hw;

    v16h a0, a1, b0, b1;
    #pragma unroll
    for (int v = 0; v < 8; ++v) {
        // A 16x32 f16 layout: VGPR v holds K = (v>>2)*16 + hi*8 + (v&3)*2 (+1)
        int kb = ((v >> 2) << 4) + (hi << 3) + ((v & 3) << 1);
        a0[2*v]   = (_Float16)W[m * 64 + kb];
        a0[2*v+1] = (_Float16)W[m * 64 + kb + 1];
        a1[2*v]   = (_Float16)W[m * 64 + kb + 32];
        a1[2*v+1] = (_Float16)W[m * 64 + kb + 33];
        // B 32x16 f16 layout: VGPR v holds K = (v>>2)*16 + hi*8 + (v&3)*2 (+1), N = lane&15
        int c0 = ((v >> 2) << 4) + (hi << 3) + ((v & 3) << 1);
        b0[2*v]   = (_Float16)X[xbase + (long)c0 * HW];
        b0[2*v+1] = (_Float16)X[xbase + (long)(c0 + 1) * HW];
        b1[2*v]   = (_Float16)X[xbase + (long)(c0 + 32) * HW];
        b1[2*v+1] = (_Float16)X[xbase + (long)(c0 + 33) * HW];
    }
    // hoist bias loads into one uniform branch (avoids per-store branching)
    float bvals[8] = {0.f, 0.f, 0.f, 0.f, 0.f, 0.f, 0.f, 0.f};
    if (bias) {
        #pragma unroll
        for (int r = 0; r < 8; ++r) bvals[r] = bias[(mt << 4) + r + (hi << 3)];
    }
    v8f acc = {};
    acc = __builtin_amdgcn_wmma_f32_16x16x32_f16(false, a0, false, b0,
                                                 (short)0, acc, false, false);
    acc = __builtin_amdgcn_wmma_f32_16x16x32_f16(false, a1, false, b1,
                                                 (short)0, acc, false, false);
    #pragma unroll
    for (int r = 0; r < 8; ++r) {
        int mrow = (mt << 4) + r + (hi << 3);     // D: VGPR r -> M = r + 8*hi
        Y[(long)(bimg * 64 + mrow) * HW + hw] = acc[r] + bvals[r];
    }
}

// ---------------------------------------------------------------------------
// 3) 3x3 local attention: softmax(q * (k_pad + relbias)) . v_pad  per channel.
// ---------------------------------------------------------------------------
__global__ void attn_kernel(const float* __restrict__ q, const float* __restrict__ k,
                            const float* __restrict__ v, const float* __restrict__ rel_h,
                            const float* __restrict__ rel_w, float* __restrict__ out) {
    int idx = blockIdx.x * blockDim.x + threadIdx.x;
    if (idx >= NTOT) return;
    int hw = idx & 16383;
    int c  = (idx >> 14) & 63;
    int h = hw >> 7, w = hw & 127;
    int bc_base = idx - hw;
    float qv = q[idx];
    float lv[9], vv[9];
    float mx = -1e30f;
    #pragma unroll
    for (int ky = 0; ky < 3; ++ky) {
        float bh = (c < 32) ? rel_h[c * 3 + ky] : 0.0f;
        #pragma unroll
        for (int kx = 0; kx < 3; ++kx) {
            float bias = (c < 32) ? bh : rel_w[(c - 32) * 3 + kx];
            int hh = h + ky - 1, ww = w + kx - 1;
            bool inb = ((unsigned)hh < 128u) && ((unsigned)ww < 128u);
            int nidx = bc_base + (hh << 7) + ww;
            float kk = (inb ? k[nidx] : 0.0f) + bias;   // pad(conv) then +bias
            float vx =  inb ? v[nidx] : 0.0f;
            int j = ky * 3 + kx;
            lv[j] = qv * kk;
            vv[j] = vx;
            mx = fmaxf(mx, lv[j]);
        }
    }
    float se = 0.f, acc = 0.f;
    #pragma unroll
    for (int j = 0; j < 9; ++j) {
        float e = __expf(lv[j] - mx);
        se += e; acc += e * vv[j];
    }
    out[idx] = acc / se;
}

// ---------------------------------------------------------------------------
// 4) tiny helpers: 64x64 matmul + collapsed bias for convv->w0->convv1 chain
// ---------------------------------------------------------------------------
__global__ void mat64_mul(float* __restrict__ Cm, const float* __restrict__ A,
                          const float* __restrict__ Bm) {
    int idx = blockIdx.x * blockDim.x + threadIdx.x;
    if (idx >= 4096) return;
    int o = idx >> 6, c = idx & 63;
    float s = 0.f;
    for (int j = 0; j < 64; ++j) s += A[o * 64 + j] * Bm[j * 64 + c];
    Cm[idx] = s;
}

__global__ void bias_combine(const float* __restrict__ w0_w, const float* __restrict__ w0_b,
                             const float* __restrict__ convv_b,
                             const float* __restrict__ convv1_w,
                             const float* __restrict__ convv1_b, float* __restrict__ bcomb) {
    int o = threadIdx.x;
    if (o >= 64) return;
    float s = convv1_b[o];
    for (int c = 0; c < 64; ++c) {
        float b1 = w0_b[c];
        for (int j = 0; j < 64; ++j) b1 += w0_w[c * 64 + j] * convv_b[j];
        s += convv1_w[o * 64 + c] * b1;
    }
    bcomb[o] = s;
}

// ---------------------------------------------------------------------------
// 5) 128-point radix-2 DIT line FFT in LDS. One block = one line (row/col).
//    Input staged global->LDS with CDNA5 async-to-LDS copies (ASYNCcnt).
//    inIm==nullptr -> zero imag input. outIm==nullptr -> real output only.
// ---------------------------------------------------------------------------
__global__ __launch_bounds__(128)
void fft128(const float* __restrict__ inRe, const float* __restrict__ inIm,
            float* __restrict__ outRe, float* __restrict__ outIm,
            int imgStrideIn, int imgStrideOut, int colMode, float dirSign, float scale) {
    __shared__ float stage_r[128], stage_i[128];
    __shared__ float sr[128], si[128];
    int line = blockIdx.x;
    int img = line >> 7, li = line & 127;
    int stride  = colMode ? 128 : 1;
    long inBase  = (long)img * imgStrideIn  + (colMode ? li : li * 128);
    long outBase = (long)img * imgStrideOut + (colMode ? li : li * 128);
    int t = threadIdx.x;

    // --- async global->LDS staging (one element per lane) ---
    // LDS flat address truncates to the 32-bit LDS offset (ISA 10.2 aperture rule)
    unsigned ldsR = (unsigned)(uintptr_t)&stage_r[t];
    unsigned long long gaR =
        (unsigned long long)(uintptr_t)(inRe + inBase + (long)t * stride);
    asm volatile("global_load_async_to_lds_b32 %0, %1, off"
                 :: "v"(ldsR), "v"(gaR) : "memory");
    if (inIm) {
        unsigned ldsI = (unsigned)(uintptr_t)&stage_i[t];
        unsigned long long gaI =
            (unsigned long long)(uintptr_t)(inIm + inBase + (long)t * stride);
        asm volatile("global_load_async_to_lds_b32 %0, %1, off"
                     :: "v"(ldsI), "v"(gaI) : "memory");
    }
    asm volatile("s_wait_asynccnt 0x0" ::: "memory");
    __syncthreads();

    // bit-reversed reorder out of the staging buffer
    unsigned r = __brev((unsigned)t) >> 25;            // 7-bit bit reversal
    sr[t] = stage_r[r];
    si[t] = inIm ? stage_i[r] : 0.0f;

    for (int len = 2, lh = 0; len <= 128; len <<= 1, ++lh) {
        __syncthreads();
        if (t < 64) {
            int half = len >> 1;
            int j = t & (half - 1);
            int grp = t >> lh;
            int i0 = grp * len + j;
            int i1 = i0 + half;
            float ang = dirSign * 6.283185307179586f * (float)j / (float)len;
            float wr = __cosf(ang), wi = __sinf(ang);
            float tr = wr * sr[i1] - wi * si[i1];
            float ti = wr * si[i1] + wi * sr[i1];
            sr[i1] = sr[i0] - tr; si[i1] = si[i0] - ti;
            sr[i0] += tr;         si[i0] += ti;
        }
    }
    __syncthreads();
    outRe[outBase + (long)t * stride] = sr[t] * scale;
    if (outIm) outIm[outBase + (long)t * stride] = si[t] * scale;
}

// ---------------------------------------------------------------------------
// 6) frequency-domain multiply: tmpB[b,o,hw] = FS32[b,hw] * wsum[o,hw] / S^2
// ---------------------------------------------------------------------------
__global__ void spec_mul(const float* __restrict__ fsr, const float* __restrict__ fsi,
                         const float* __restrict__ wr, const float* __restrict__ wi,
                         float* __restrict__ outr, float* __restrict__ outi) {
    int idx = blockIdx.x * blockDim.x + threadIdx.x;
    if (idx >= NTOT) return;
    int hw = idx & 16383;
    int o  = (idx >> 14) & 63;
    int b  = idx >> 20;
    float ar = fsr[(b << 14) + hw], ai = fsi[(b << 14) + hw];
    float br = wr[(o << 14) + hw],  bi = wi[(o << 14) + hw];
    const float sc = 1.0f / 16384.0f;                  // ortho norms of fft2+ifft2
    outr[idx] = (ar * br - ai * bi) * sc;
    outi[idx] = (ar * bi + ai * br) * sc;
}

// ---------------------------------------------------------------------------
// 7) deterministic two-stage MSE reduction
// ---------------------------------------------------------------------------
__global__ void loss_partial(const float* __restrict__ out1, const float* __restrict__ out2,
                             const float* __restrict__ ft, const float* __restrict__ rate1,
                             const float* __restrict__ rate2, float* __restrict__ partials) {
    __shared__ float sdata[256];
    float r1 = rate1[0], r2 = rate2[0];
    float s = 0.f;
    for (int e = blockIdx.x * 256 + threadIdx.x; e < NTOT; e += 1024 * 256) {
        int c = e & 63;
        int w = (e >> 6) & 127;
        int h = (e >> 13) & 127;
        int b = e >> 20;
        int ci = ((b * 64 + c) << 14) + (h << 7) + w;  // [B,C,H,W]
        float mk = r1 * out1[ci] + r2 * out2[ci];
        float d = mk - ft[e];                          // ft already [B,H,W,C]
        s += d * d;
    }
    sdata[threadIdx.x] = s;
    __syncthreads();
    for (int st = 128; st > 0; st >>= 1) {
        if (threadIdx.x < st) sdata[threadIdx.x] += sdata[threadIdx.x + st];
        __syncthreads();
    }
    if (threadIdx.x == 0) partials[blockIdx.x] = sdata[0];
}

__global__ void loss_final(const float* __restrict__ partials, float* __restrict__ out) {
    __shared__ float sdata[256];
    float s = 0.f;
    for (int i = threadIdx.x; i < 1024; i += 256) s += partials[i];
    sdata[threadIdx.x] = s;
    __syncthreads();
    for (int st = 128; st > 0; st >>= 1) {
        if (threadIdx.x < st) sdata[threadIdx.x] += sdata[threadIdx.x + st];
        __syncthreads();
    }
    if (threadIdx.x == 0) out[0] = sdata[0] / (float)NTOT;
}

// ---------------------------------------------------------------------------
extern "C" void kernel_launch(void* const* d_in, const int* in_sizes, int n_in,
                              void* d_out, int out_size, void* d_ws, size_t ws_size,
                              hipStream_t stream) {
    const float* fs       = (const float*)d_in[0];
    const float* ft       = (const float*)d_in[1];
    const float* w_q      = (const float*)d_in[2];
    const float* w_k      = (const float*)d_in[3];
    const float* w_v      = (const float*)d_in[4];
    const float* rel_h    = (const float*)d_in[5];
    const float* rel_w    = (const float*)d_in[6];
    const float* w1_real  = (const float*)d_in[7];
    const float* w1_imag  = (const float*)d_in[8];
    const float* w0_w     = (const float*)d_in[9];
    const float* w0_b     = (const float*)d_in[10];
    const float* convv_w  = (const float*)d_in[11];
    const float* convv_b  = (const float*)d_in[12];
    const float* convv1_w = (const float*)d_in[13];
    const float* convv1_b = (const float*)d_in[14];
    const float* rate1    = (const float*)d_in[15];
    const float* rate2    = (const float*)d_in[16];
    float* outS = (float*)d_out;

    float* ws = (float*)d_ws;
    // workspace layout (floats); buffers reused once their producers are consumed
    float* qbuf   = ws;                       // 2,097,152  (later reused as out1)
    float* kbuf   = qbuf   + NTOT;            // 2,097,152  (later reused as out2)
    float* vbuf   = kbuf   + NTOT;            // 2,097,152  (later reused as tmpB_re)
    float* outbuf = vbuf   + NTOT;            // 2,097,152
    float* wsum_r = outbuf + NTOT;            // 1,048,576
    float* wsum_i = wsum_r + C * HW;          // 1,048,576
    float* tmpA_r = wsum_i + C * HW;          // 32,768  (becomes FS32 re)
    float* tmpA_i = tmpA_r + Bq * HW;         // 32,768  (becomes FS32 im)
    float* tmpB_i = tmpA_i + Bq * HW;         // 2,097,152
    float* T1     = tmpB_i + NTOT;            // 4,096
    float* Mcomb  = T1     + 4096;            // 4,096
    float* bcomb  = Mcomb  + 4096;            // 64
    float* parts  = bcomb  + 64;              // 1,024
    float* tmpB_r = vbuf;                     // reuse
    float* out1   = qbuf;                     // reuse
    float* out2   = kbuf;                     // reuse

    dim3 gemmGrid(2048, 4), gemmBlk(32);

    // 1) dominant streaming reduction over weights1 (512 MB)
    wsum_kernel<<<1024, 256, 0, stream>>>((const f4*)w1_real, (const f4*)w1_imag,
                                          (f4*)wsum_r, (f4*)wsum_i);

    // 2) q,k,v 1x1 convs via WMMA
    gemm64_wmma<<<gemmGrid, gemmBlk, 0, stream>>>(w_q, nullptr, fs, qbuf);
    gemm64_wmma<<<gemmGrid, gemmBlk, 0, stream>>>(w_k, nullptr, fs, kbuf);
    gemm64_wmma<<<gemmGrid, gemmBlk, 0, stream>>>(w_v, nullptr, fs, vbuf);

    // 3) local attention
    attn_kernel<<<(NTOT + 255) / 256, 256, 0, stream>>>(qbuf, kbuf, vbuf,
                                                        rel_h, rel_w, outbuf);

    // 4) collapse convv -> w0 -> convv1 into one matrix + bias, then WMMA GEMM
    mat64_mul<<<16, 256, 0, stream>>>(T1, w0_w, convv_w);        // T1 = w0 @ convv
    mat64_mul<<<16, 256, 0, stream>>>(Mcomb, convv1_w, T1);      // M  = convv1 @ T1
    bias_combine<<<1, 64, 0, stream>>>(w0_w, w0_b, convv_b, convv1_w, convv1_b, bcomb);
    gemm64_wmma<<<gemmGrid, gemmBlk, 0, stream>>>(Mcomb, bcomb, outbuf, out2);

    // 5) spectral branch:
    //    forward fft2 of out[:,32] (rows then cols), freq multiply, inverse fft2
    fft128<<<Bq * S, 128, 0, stream>>>(outbuf + 32 * HW, nullptr, tmpA_r, tmpA_i,
                                       64 * HW, HW, /*col=*/0, -1.0f, 1.0f);
    fft128<<<Bq * S, 128, 0, stream>>>(tmpA_r, tmpA_i, tmpA_r, tmpA_i,
                                       HW, HW, /*col=*/1, -1.0f, 1.0f);
    spec_mul<<<(NTOT + 255) / 256, 256, 0, stream>>>(tmpA_r, tmpA_i, wsum_r, wsum_i,
                                                     tmpB_r, tmpB_i);
    fft128<<<Bq * C * S, 128, 0, stream>>>(tmpB_r, tmpB_i, tmpB_r, tmpB_i,
                                           HW, HW, /*col=*/0, +1.0f, 1.0f);
    fft128<<<Bq * C * S, 128, 0, stream>>>(tmpB_r, tmpB_i, out1, nullptr,
                                           HW, HW, /*col=*/1, +1.0f, 1.0f);

    // 6) MSE loss, deterministic two-stage reduction
    loss_partial<<<1024, 256, 0, stream>>>(out1, out2, ft, rate1, rate2, parts);
    loss_final<<<1, 256, 0, stream>>>(parts, outS);
}
